// MemoryTripletK_reuse_34102040330865
// MI455X (gfx1250) — matrix operand order repaired and checked by reading, not compiled, not run on hardware
//
#include <hip/hip_runtime.h>

// ---- problem constants ----
#define N_ROWS 128
#define D_DIM  2048
#define M_ROWS 16384
#define C_CLS  65
#define CHW    150528
#define NPAD   80          // pad C 65 -> 80 (5 N-tiles of 16)
#define KTOP   5
#define KSPLIT 56          // K-split for G GEMM
#define KSTEPS 84          // 56 * 84 * 32 == 150528
#define KCHUNK 128         // bf16 elements of K staged per LDS chunk
#define INF_F  1e30f

typedef __bf16 bf16_t;
typedef __attribute__((ext_vector_type(16))) __bf16 v16bf;
typedef __attribute__((ext_vector_type(8)))  __bf16 v8bf;
typedef __attribute__((ext_vector_type(8)))  float  v8f;

__device__ inline bf16_t to_bf16(float f) {
  unsigned u = __builtin_bit_cast(unsigned, f);
  unsigned r = u + 0x7FFFu + ((u >> 16) & 1u);           // RNE
  unsigned short h = (unsigned short)(r >> 16);
  return __builtin_bit_cast(bf16_t, h);
}

// ---- CDNA5 async global->LDS staging (ASYNCcnt path) ----
__device__ inline void async_b128_to_lds(unsigned lds_byte_off, const void* gaddr) {
  // VDST = VGPR holding LDS byte address, VADDR = 64-bit global address, no SADDR
  asm volatile("global_load_async_to_lds_b128 %0, %1, off"
               :: "v"(lds_byte_off), "v"(gaddr) : "memory");
}
__device__ inline void async_wait0() {
#if __has_builtin(__builtin_amdgcn_s_wait_asynccnt)
  __builtin_amdgcn_s_wait_asynccnt(0);
#else
  asm volatile("s_wait_asynccnt 0x0" ::: "memory");
#endif
}
__device__ inline unsigned lds_off32(const void* p) {
  // flat shared pointer = {aperture_hi32, lds_offset}; low 32 bits are the DS offset
  return (unsigned)(size_t)p;
}

// ---------------- pseudo = argmax(target_row_t, axis=1) ----------------
__global__ void k_pseudo(const float* __restrict__ trt, int* __restrict__ pseudo) {
  int j = blockIdx.x * blockDim.x + threadIdx.x;
  if (j >= M_ROWS) return;
  const float* r = trt + (size_t)j * C_CLS;
  float bv = r[0]; int bi = 0;
  for (int c = 1; c < C_CLS; c++) { float v = r[c]; if (v > bv) { bv = v; bi = c; } }
  pseudo[j] = bi;
}

// ---------------- per-class stats (deterministic, no atomics) ----------------
__global__ void k_cls(const int* __restrict__ pseudo, const float* __restrict__ wstate,
                      float* __restrict__ cls_sum, float* __restrict__ cls_cnt) {
  int c = blockIdx.x, t = threadIdx.x;
  __shared__ float r1[256], r2[256];
  float s = 0.f, n = 0.f;
  for (int j = t; j < M_ROWS; j += 256)
    if (pseudo[j] == c) { s += wstate[j]; n += 1.0f; }
  r1[t] = s; r2[t] = n; __syncthreads();
  for (int o = 128; o > 0; o >>= 1) { if (t < o) { r1[t] += r1[t+o]; r2[t] += r2[t+o]; } __syncthreads(); }
  if (t == 0) { cls_sum[c] = r1[0]; cls_cnt[c] = r2[0]; }
}

// ---------------- row-normalize + f32->bf16 (row length D_DIM) ----------------
__global__ void k_norm(const float* __restrict__ x, bf16_t* __restrict__ xb,
                       float* __restrict__ snorm) {
  int i = blockIdx.x, t = threadIdx.x;
  __shared__ float red[256];
  const float* row = x + (size_t)i * D_DIM;
  float s = 0.f;
  for (int d = t; d < D_DIM; d += 256) { float v = row[d]; s += v * v; }
  red[t] = s; __syncthreads();
  for (int o = 128; o > 0; o >>= 1) { if (t < o) red[t] += red[t+o]; __syncthreads(); }
  float inv = 1.0f / fmaxf(sqrtf(red[0]), 1e-12f);
  __syncthreads();
  bf16_t* orow = xb + (size_t)i * D_DIM;
  float s2 = 0.f;
  for (int d = t; d < D_DIM; d += 256) { float v = row[d] * inv; orow[d] = to_bf16(v); s2 += v * v; }
  red[t] = s2; __syncthreads();
  for (int o = 128; o > 0; o >>= 1) { if (t < o) red[t] += red[t+o]; __syncthreads(); }
  if (t == 0) snorm[i] = red[0];
}

// ---------------- sim GEMM: an(128xD) x bn(16384xD)^T via WMMA bf16 ----------------
// One block per 16-col N-tile; its 8 waves cover the 8 M-tiles and share the B tile,
// which is double-buffer staged to LDS with async global->LDS loads (8x less L2 traffic).
__global__ void __launch_bounds__(256) k_sim(const bf16_t* __restrict__ an,
                                             const bf16_t* __restrict__ bn,
                                             const float* __restrict__ sa,
                                             const float* __restrict__ sb,
                                             float* __restrict__ sim) {
  __shared__ __align__(16) bf16_t sB[2][16 * KCHUNK];   // 2 x 4KB
  int t = threadIdx.x;
  int lane = t & 31;
  int wave = t >> 5;                 // M-tile index (0..7)
  int n0 = blockIdx.x * 16;          // 1024 blocks
  int m0 = wave * 16;
  int l = lane & 15, hi = lane >> 4;

  // staging role: 256 threads cover 16 rows x 128 bf16 (16B each)
  int sr = t >> 4, sc = (t & 15) * 8;
  const bf16_t* gsrc = bn + (size_t)(n0 + sr) * D_DIM + sc;
  unsigned lbase0 = lds_off32(&sB[0][0]);
  unsigned lbase1 = lds_off32(&sB[1][0]);
  unsigned lofs = (unsigned)(sr * KCHUNK + sc) * 2u;

  const bf16_t* pa = an + (size_t)(m0 + l) * D_DIM + hi * 16;  // A: lane=M, contiguous K halves
  v8f c = {0.f,0.f,0.f,0.f,0.f,0.f,0.f,0.f};

  async_b128_to_lds(lbase0 + lofs, (const void*)gsrc);
  async_wait0();
  __syncthreads();

  const int NCH = D_DIM / KCHUNK;   // 16 chunks
  for (int ch = 0; ch < NCH; ch++) {
    if (ch + 1 < NCH)
      async_b128_to_lds(((ch + 1) & 1 ? lbase1 : lbase0) + lofs,
                        (const void*)(gsrc + (ch + 1) * KCHUNK));
    const bf16_t* bbuf = &sB[ch & 1][0];
    int kc = ch * KCHUNK;
    #pragma unroll
    for (int s = 0; s < KCHUNK / 32; s++) {             // 4 WMMA steps per chunk
      int kk = kc + s * 32;
      v8bf a0 = *(const v8bf*)(pa + kk);
      v8bf a1 = *(const v8bf*)(pa + kk + 8);
      v16bf av = __builtin_shufflevector(a0, a1, 0,1,2,3,4,5,6,7,8,9,10,11,12,13,14,15);
      v16bf bv = *(const v16bf*)(bbuf + l * KCHUNK + s * 32 + hi * 16); // B: lane=N
      c = __builtin_amdgcn_wmma_f32_16x16x32_bf16(false, av, false, bv, (short)0, c, false, false);
    }
    async_wait0();
    __syncthreads();
  }

  int n = n0 + l;
  float sbn = sb[n];
  #pragma unroll
  for (int r = 0; r < 8; r++) {
    int m = m0 + hi * 8 + r;                      // D layout: VGPR r -> M = r + 8*hi, N = lane&15
    float v = sa[m] + sbn - 2.0f * c[r];
    sim[(size_t)m * M_ROWS + n] = sqrtf(fmaxf(v, 0.0f));
  }
}

__global__ void k_diag(float* sim, const int* __restrict__ idx_t) {
  int i = threadIdx.x;
  if (i < N_ROWS) sim[(size_t)i * M_ROWS + idx_t[i]] = 10.0f;
}

// ---------------- per-row top-K selection (4 masked lists) ----------------
__global__ void k_select(const float* __restrict__ sim, const int* __restrict__ pseudo,
                         const int* __restrict__ tcol, const float* __restrict__ wstate,
                         const float* __restrict__ cls_sum, const float* __restrict__ cls_cnt,
                         int* __restrict__ pos_idx, int* __restrict__ neg_idx,
                         int* __restrict__ valid, int* __restrict__ flag,
                         float* __restrict__ dout) {
  int i = blockIdx.x, t = threadIdx.x;
  __shared__ float rv[256]; __shared__ int ri[256];
  __shared__ float th[2];  __shared__ int chosen[4][KTOP];
  __shared__ float sums[2]; __shared__ int cnts[2];
  int tc = tcol[i];
  if (t == 0) {
    float tot = 0.f;
    for (int c = 0; c < C_CLS; c++) tot += cls_sum[c];
    float cs = cls_sum[tc], cc = cls_cnt[tc];
    th[0] = cs / cc;
    th[1] = (tot - cs) / ((float)M_ROWS - cc);
    sums[0] = 0.f; sums[1] = 0.f;
  }
  __syncthreads();
  float yp = th[0], yn = th[1];
  const float* srow = sim + (size_t)i * M_ROWS;
  int c2p = 0, c2n = 0;
  for (int j = t; j < M_ROWS; j += 256) {
    bool pos = (pseudo[j] == tc);
    float w = wstate[j];
    if (pos && w >= yp) c2p++;
    if (!pos && w >= yn) c2n++;
  }
  ri[t] = c2p; __syncthreads();
  for (int o = 128; o > 0; o >>= 1) { if (t < o) ri[t] += ri[t+o]; __syncthreads(); }
  if (t == 0) cnts[0] = ri[0];
  __syncthreads();
  ri[t] = c2n; __syncthreads();
  for (int o = 128; o > 0; o >>= 1) { if (t < o) ri[t] += ri[t+o]; __syncthreads(); }
  if (t == 0) cnts[1] = ri[0];
  __syncthreads();
  // 5 rounds; lists: 0=pos-min, 1=neg-min, 2=pmask2-max(idx), 3=nmask2-min(idx)
  for (int r = 0; r < KTOP; r++) {
    float bv[4] = {INF_F, INF_F, -INF_F, INF_F};
    int   bi[4] = {0x7fffffff, 0x7fffffff, 0x7fffffff, 0x7fffffff};
    for (int j = t; j < M_ROWS; j += 256) {
      float s = srow[j];
      bool pos = (pseudo[j] == tc);
      float w = wstate[j];
      bool ex0=false, ex1=false, ex2=false, ex3=false;
      for (int q = 0; q < r; q++) {
        ex0 |= (chosen[0][q] == j); ex1 |= (chosen[1][q] == j);
        ex2 |= (chosen[2][q] == j); ex3 |= (chosen[3][q] == j);
      }
      if (pos && !ex0)             { if (s < bv[0] || (s == bv[0] && j < bi[0])) { bv[0]=s; bi[0]=j; } }
      if (!pos && !ex1)            { if (s < bv[1] || (s == bv[1] && j < bi[1])) { bv[1]=s; bi[1]=j; } }
      if (pos && w >= yp && !ex2)  { if (s > bv[2] || (s == bv[2] && j < bi[2])) { bv[2]=s; bi[2]=j; } }
      if (!pos && w >= yn && !ex3) { if (s < bv[3] || (s == bv[3] && j < bi[3])) { bv[3]=s; bi[3]=j; } }
    }
    for (int list = 0; list < 4; list++) {
      rv[t] = bv[list]; ri[t] = bi[list]; __syncthreads();
      for (int o = 128; o > 0; o >>= 1) {
        if (t < o) {
          float ov = rv[t+o]; int oi = ri[t+o];
          bool take = (list == 2) ? ((ov > rv[t]) || (ov == rv[t] && oi < ri[t]))
                                  : ((ov < rv[t]) || (ov == rv[t] && oi < ri[t]));
          if (take) { rv[t] = ov; ri[t] = oi; }
        }
        __syncthreads();
      }
      if (t == 0) {
        int idx = ri[0]; if (idx < 0 || idx >= M_ROWS) idx = 0;
        chosen[list][r] = idx;
        if (list == 0) sums[0] += rv[0];
        if (list == 1) sums[1] += rv[0];
      }
      __syncthreads();
    }
  }
  if (t == 0) {
    float Wt = sums[1] / sums[0];
    dout[2 + i] = Wt;                               // W_tt output
    flag[i]  = (Wt >= yp) ? 1 : 0;
    valid[i] = (cnts[0] >= KTOP && cnts[1] >= KTOP) ? 1 : 0;
    for (int r = 0; r < KTOP; r++) {
      pos_idx[i * KTOP + r] = chosen[2][r];
      neg_idx[i * KTOP + r] = chosen[3][r];
    }
  }
}

// ---------------- triplet distances ----------------
__global__ void k_trip(const float* __restrict__ icol, const float* __restrict__ irow,
                       const int* __restrict__ pos_idx, const int* __restrict__ neg_idx,
                       const int* __restrict__ valid, const float* __restrict__ weight,
                       float* __restrict__ trip_w, float* __restrict__ weff) {
  int i = blockIdx.x, t = threadIdx.x;
  __shared__ float red[256];
  const float* a = icol + (size_t)i * D_DIM;
  const float* P[KTOP]; const float* Q[KTOP];
  for (int k = 0; k < KTOP; k++) {
    P[k] = irow + (size_t)pos_idx[i * KTOP + k] * D_DIM;
    Q[k] = irow + (size_t)neg_idx[i * KTOP + k] * D_DIM;
  }
  float ap[KTOP] = {0,0,0,0,0}, an_[KTOP] = {0,0,0,0,0};
  for (int d = t; d < D_DIM; d += 256) {
    float av = a[d];
    #pragma unroll
    for (int k = 0; k < KTOP; k++) {
      float dp = av - P[k][d] + 1e-6f; ap[k]  += dp * dp;
      float dn = av - Q[k][d] + 1e-6f; an_[k] += dn * dn;
    }
  }
  float dap[KTOP], dan[KTOP];
  for (int k = 0; k < KTOP; k++) {
    red[t] = ap[k]; __syncthreads();
    for (int o = 128; o > 0; o >>= 1) { if (t < o) red[t] += red[t+o]; __syncthreads(); }
    dap[k] = red[0]; __syncthreads();
    red[t] = an_[k]; __syncthreads();
    for (int o = 128; o > 0; o >>= 1) { if (t < o) red[t] += red[t+o]; __syncthreads(); }
    dan[k] = red[0]; __syncthreads();
  }
  if (t == 0) {
    float s = 0.f;
    for (int k = 0; k < KTOP; k++) s += fmaxf(sqrtf(dap[k]) - sqrtf(dan[k]) + 0.3f, 0.0f);
    float w = weight[i] * (valid[i] ? 1.0f : 0.0f);
    trip_w[i] = (s * 0.2f) * w;
    weff[i] = w;
  }
}

// ---------------- MT19937 (numpy legacy) ----------------
__device__ unsigned mt_next(unsigned* mt, int* mti) {
  if (*mti >= 624) {
    for (int k = 0; k < 624; k++) {
      unsigned y = (mt[k] & 0x80000000u) | (mt[(k + 1) % 624] & 0x7fffffffu);
      unsigned v = mt[(k + 397) % 624] ^ (y >> 1);
      if (y & 1u) v ^= 0x9908b0dfu;
      mt[k] = v;
    }
    *mti = 0;
  }
  unsigned y = mt[(*mti)++];
  y ^= y >> 11; y ^= (y << 7) & 0x9d2c5680u; y ^= (y << 15) & 0xefc60000u; y ^= y >> 18;
  return y;
}
__device__ double mt_double(unsigned* mt, int* mti) {
  unsigned a = mt_next(mt, mti) >> 5, b = mt_next(mt, mti) >> 6;
  return (a * 67108864.0 + b) / 9007199254740992.0;
}

// ---------------- loss reduction + sel/flag + RNG ----------------
__global__ void k_final(const float* __restrict__ trip_w, const float* __restrict__ weff,
                        const int* __restrict__ flag, float* __restrict__ dout,
                        int* __restrict__ sel, int* __restrict__ shuf,
                        float* __restrict__ lam, int* __restrict__ lenp) {
  int t = threadIdx.x;
  __shared__ float r1[128], r2[128];
  __shared__ unsigned mt[624];
  r1[t] = trip_w[t]; r2[t] = weff[t]; __syncthreads();
  for (int o = 64; o > 0; o >>= 1) { if (t < o) { r1[t] += r1[t+o]; r2[t] += r2[t+o]; } __syncthreads(); }
  if (t == 0) {
    dout[0] = r1[0] / (1e-5f + r2[0]);
    int len = 0;
    for (int i = 0; i < N_ROWS; i++) if (flag[i]) sel[len++] = i;
    *lenp = len;
    if (len > 1) {
      int mti;
      mt[0] = 1u;                                     // np.random.seed(1)
      for (mti = 1; mti < 624; mti++)
        mt[mti] = 1812433253u * (mt[mti-1] ^ (mt[mti-1] >> 30)) + (unsigned)mti;
      for (int b = 0; b < len; b++) {                 // beta(1,1) via Johnk
        double U, V;
        do { U = mt_double(mt, &mti); V = mt_double(mt, &mti); } while (U + V > 1.0);
        double s = U + V;
        lam[b] = (s > 0.0) ? (float)(U / s) : 0.5f;
      }
      for (int b = 0; b < len; b++) shuf[b] = b;      // Fisher-Yates (rk_interval)
      for (int i2 = len - 1; i2 > 0; i2--) {
        unsigned mask = (unsigned)i2;
        mask |= mask >> 1; mask |= mask >> 2; mask |= mask >> 4; mask |= mask >> 8; mask |= mask >> 16;
        unsigned v;
        do { v = mt_next(mt, &mti) & mask; } while (v > (unsigned)i2);
        int tmp = shuf[i2]; shuf[i2] = shuf[v]; shuf[v] = tmp;
      }
    }
  }
}

// ---------------- bf16 conversions for the G GEMM ----------------
__global__ void k_conv_img(const float* __restrict__ img, bf16_t* __restrict__ out) {
  size_t i = ((size_t)blockIdx.x * blockDim.x + threadIdx.x) * 4;
  #pragma unroll
  for (int q = 0; q < 4; q++) out[i + q] = to_bf16(img[i + q]);
}

__global__ void k_conv_w(const float* __restrict__ wm, bf16_t* __restrict__ wt) {
  __shared__ bf16_t tile[32][NPAD + 8];
  int k0 = blockIdx.x * 32, t = threadIdx.x;
  for (int e = t; e < 32 * C_CLS; e += 256) {
    int kk = e / C_CLS, n = e % C_CLS;
    tile[kk][n] = to_bf16(wm[(size_t)(k0 + kk) * C_CLS + n]);
  }
  for (int e = t; e < 32 * (NPAD - C_CLS); e += 256) {
    int kk = e / (NPAD - C_CLS), n = C_CLS + e % (NPAD - C_CLS);
    tile[kk][n] = to_bf16(0.0f);
  }
  __syncthreads();
  for (int e = t; e < NPAD * 32; e += 256) {
    int n = e / 32, kk = e % 32;
    wt[(size_t)n * CHW + k0 + kk] = tile[kk][n];
  }
}

// ---------------- G = image * W_model via WMMA bf16, K-split partials ----------------
// Block = (kc, ntile); its 8 waves cover the 8 M-tiles and share the async-staged B tile.
__global__ void __launch_bounds__(256) k_gemmG(const bf16_t* __restrict__ A,
                                               const bf16_t* __restrict__ B,
                                               float* __restrict__ Gpart) {
  __shared__ __align__(16) bf16_t sB[2][16 * KCHUNK];
  int t = threadIdx.x;
  int lane = t & 31;
  int wave = t >> 5;                       // M-tile (0..7)
  int kc = blockIdx.x / 5;                 // 280 blocks: 56 kc x 5 ntile
  int ntile = blockIdx.x % 5;
  int m0 = wave * 16, n0 = ntile * 16;
  int l = lane & 15, hi = lane >> 4;
  size_t kbase = (size_t)kc * KSTEPS * 32; // 2688 K per chunk-set

  int sr = t >> 4, sc = (t & 15) * 8;
  const bf16_t* gsrc = B + (size_t)(n0 + sr) * CHW + kbase + sc;
  unsigned lbase0 = lds_off32(&sB[0][0]);
  unsigned lbase1 = lds_off32(&sB[1][0]);
  unsigned lofs = (unsigned)(sr * KCHUNK + sc) * 2u;

  const bf16_t* pa = A + (size_t)(m0 + l) * CHW + kbase + hi * 16;
  v8f c = {0.f,0.f,0.f,0.f,0.f,0.f,0.f,0.f};

  async_b128_to_lds(lbase0 + lofs, (const void*)gsrc);
  async_wait0();
  __syncthreads();

  const int NCH = (KSTEPS * 32) / KCHUNK;  // 21 chunks
  for (int ch = 0; ch < NCH; ch++) {
    if (ch + 1 < NCH)
      async_b128_to_lds(((ch + 1) & 1 ? lbase1 : lbase0) + lofs,
                        (const void*)(gsrc + (ch + 1) * KCHUNK));
    const bf16_t* bbuf = &sB[ch & 1][0];
    int kc0 = ch * KCHUNK;
    #pragma unroll
    for (int s = 0; s < KCHUNK / 32; s++) {
      int kk = kc0 + s * 32;
      v8bf a0 = *(const v8bf*)(pa + kk);
      v8bf a1 = *(const v8bf*)(pa + kk + 8);
      v16bf av = __builtin_shufflevector(a0, a1, 0,1,2,3,4,5,6,7,8,9,10,11,12,13,14,15);
      v16bf bv = *(const v16bf*)(bbuf + l * KCHUNK + s * 32 + hi * 16);
      c = __builtin_amdgcn_wmma_f32_16x16x32_bf16(false, av, false, bv, (short)0, c, false, false);
    }
    async_wait0();
    __syncthreads();
  }

  float* g = Gpart + (size_t)kc * (N_ROWS * NPAD);
  int n = n0 + l;
  #pragma unroll
  for (int r = 0; r < 8; r++) {
    int m = m0 + hi * 8 + r;
    g[m * NPAD + n] = c[r];
  }
}

__global__ void k_redG(const float* __restrict__ Gpart, float* __restrict__ G) {
  int i = blockIdx.x * blockDim.x + threadIdx.x;   // 10240 threads
  float s = 0.f;
  for (int c = 0; c < KSPLIT; c++) s += Gpart[(size_t)c * (N_ROWS * NPAD) + i];
  G[i] = s;
}

// ---------------- mixup softmax cross-entropy ----------------
__global__ void k_reloss(const float* __restrict__ G, const int* __restrict__ sel,
                         const int* __restrict__ shuf, const float* __restrict__ lam,
                         const int* __restrict__ lenp, const int* __restrict__ tcol,
                         float* __restrict__ dout) {
  int t = threadIdx.x;
  __shared__ float red[128];
  __shared__ float sexp[C_CLS];
  __shared__ float total;
  int len = *lenp;
  if (len <= 1) { if (t == 0) dout[1] = 0.0f; return; }
  if (t == 0) total = 0.0f;
  __syncthreads();
  for (int b = 0; b < len; b++) {
    int s1 = sel[b], s2 = sel[shuf[b]];
    float l = lam[b];
    float lg = (t < C_CLS) ? (l * G[s1 * NPAD + t] + (1.0f - l) * G[s2 * NPAD + t]) : -INF_F;
    red[t] = lg; __syncthreads();
    for (int o = 64; o > 0; o >>= 1) { if (t < o) red[t] = fmaxf(red[t], red[t+o]); __syncthreads(); }
    float mx = red[0]; __syncthreads();
    float e = (t < C_CLS) ? expf(lg - mx) : 0.0f;
    if (t < C_CLS) sexp[t] = e;
    red[t] = e; __syncthreads();
    for (int o = 64; o > 0; o >>= 1) { if (t < o) red[t] += red[t+o]; __syncthreads(); }
    if (t == 0) {
      float ssum = red[0];
      float p1 = sexp[tcol[s1]] / ssum, p2 = sexp[tcol[s2]] / ssum;
      total += -(l * logf(p1 + 1e-5f) + (1.0f - l) * logf(p2 + 1e-5f));
    }
    __syncthreads();
  }
  if (t == 0) dout[1] = total / (float)len;
}

extern "C" void kernel_launch(void* const* d_in, const int* in_sizes, int n_in,
                              void* d_out, int out_size, void* d_ws, size_t ws_size,
                              hipStream_t stream) {
  (void)in_sizes; (void)n_in; (void)out_size; (void)ws_size;
  const float* inputs_col  = (const float*)d_in[0];
  const int*   targets_col = (const int*)  d_in[1];
  const int*   idx_t       = (const int*)  d_in[2];
  const float* inputs_row  = (const float*)d_in[3];
  const float* target_row  = (const float*)d_in[4];
  const float* weight      = (const float*)d_in[5];
  const float* image       = (const float*)d_in[6];
  /* d_in[7] = p_t (unused by reference) */
  const float* wstate      = (const float*)d_in[8];
  const float* wmodel      = (const float*)d_in[9];
  float* dout = (float*)d_out;

  char* ws = (char*)d_ws;
  size_t off = 0;
  auto alloc = [&](size_t bytes) -> char* {
    char* p = ws + off; off = (off + bytes + 255) & ~(size_t)255; return p;
  };
  // region reused: bn_bf16 (67MB) first, then img_bf16 (38.5MB) + wt_bf16 (24MB)
  char* region = alloc((size_t)M_ROWS * D_DIM * 2);
  bf16_t* bn   = (bf16_t*)region;
  bf16_t* imgb = (bf16_t*)region;
  bf16_t* wtb  = (bf16_t*)(region + (size_t)N_ROWS * CHW * 2);
  float*  sim  = (float*) alloc((size_t)N_ROWS * M_ROWS * 4);
  bf16_t* an   = (bf16_t*)alloc((size_t)N_ROWS * D_DIM * 2);
  float* sa      = (float*)alloc(N_ROWS * 4);
  float* sb      = (float*)alloc(M_ROWS * 4);
  int*   pseudo  = (int*)  alloc(M_ROWS * 4);
  float* cls_sum = (float*)alloc(C_CLS * 4);
  float* cls_cnt = (float*)alloc(C_CLS * 4);
  int*   pos_idx = (int*)  alloc(N_ROWS * KTOP * 4);
  int*   neg_idx = (int*)  alloc(N_ROWS * KTOP * 4);
  int*   valid   = (int*)  alloc(N_ROWS * 4);
  int*   flag    = (int*)  alloc(N_ROWS * 4);
  float* trip_w  = (float*)alloc(N_ROWS * 4);
  float* weff    = (float*)alloc(N_ROWS * 4);
  int*   sel     = (int*)  alloc(N_ROWS * 4);
  int*   shuf    = (int*)  alloc(N_ROWS * 4);
  float* lam     = (float*)alloc(N_ROWS * 4);
  int*   lenp    = (int*)  alloc(4);
  float* Gpart   = (float*)alloc((size_t)KSPLIT * N_ROWS * NPAD * 4);
  float* G       = (float*)alloc((size_t)N_ROWS * NPAD * 4);

  // phase 1: sim path (uses bn region)
  k_pseudo<<<M_ROWS / 256, 256, 0, stream>>>(target_row, pseudo);
  k_cls<<<C_CLS, 256, 0, stream>>>(pseudo, wstate, cls_sum, cls_cnt);
  k_norm<<<N_ROWS, 256, 0, stream>>>(inputs_col, an, sa);
  k_norm<<<M_ROWS, 256, 0, stream>>>(inputs_row, bn, sb);
  k_sim<<<M_ROWS / 16, 256, 0, stream>>>(an, bn, sa, sb, sim);
  k_diag<<<1, N_ROWS, 0, stream>>>(sim, idx_t);
  k_select<<<N_ROWS, 256, 0, stream>>>(sim, pseudo, targets_col, wstate, cls_sum, cls_cnt,
                                       pos_idx, neg_idx, valid, flag, dout);
  k_trip<<<N_ROWS, 256, 0, stream>>>(inputs_col, inputs_row, pos_idx, neg_idx, valid, weight,
                                     trip_w, weff);
  k_final<<<1, 128, 0, stream>>>(trip_w, weff, flag, dout, sel, shuf, lam, lenp);
  // phase 2: G GEMM path (reuses bn region after k_sim completed in stream order)
  k_conv_img<<<(N_ROWS * CHW) / (256 * 4), 256, 0, stream>>>(image, imgb);
  k_conv_w<<<CHW / 32, 256, 0, stream>>>(wmodel, wtb);
  k_gemmG<<<KSPLIT * 5, 256, 0, stream>>>(imgb, wtb, Gpart);
  k_redG<<<(N_ROWS * NPAD) / 256, 256, 0, stream>>>(Gpart, G);
  k_reloss<<<1, 128, 0, stream>>>(G, sel, shuf, lam, lenp, targets_col, dout);
}
